// GATCNNMF_40673340293245
// MI455X (gfx1250) — matrix operand order repaired and checked by reading, not compiled, not run on hardware
//
#include <hip/hip_runtime.h>

// ---------------- problem constants (from reference) ----------------
#define R_  600
#define D_  240
#define N_  840          // R + D
#define H_  8
#define C_  64
#define E_  20000
#define NE_ (E_ + N_)    // edges incl. self-loops = 20840
#define HC_ 512          // H*C
#define F_  1242         // 6*(64+61+49+33)
#define NEG_SLOPE 0.2f

typedef __attribute__((ext_vector_type(16))) _Float16 v16h;
typedef __attribute__((ext_vector_type(8)))  _Float16 v8h;
typedef __attribute__((ext_vector_type(2)))  _Float16 h2;
typedef __attribute__((ext_vector_type(8)))  float    v8f;
typedef __attribute__((ext_vector_type(2)))  float    f2v;   // clang-native vector

// =====================================================================
// WMMA GEMM: C[M x Nc] = act( A[M x K] @ B[K x Nc] + bias )
// fp32 in/out, f16 WMMA with f32 accumulate.
// Block: 256 thr = 8 waves. Block tile 64(M) x 64(N), K step 32.
// Wave tile 16x32 -> 2 x v_wmma_f32_16x16x32_f16 per K step.
// Main K loop has no K-bounds checks (K & ~31); a single clamped epilogue
// step handles the remainder. M/N edge tiles use clamped vector loads +
// selects (no divergent branches anywhere in the hot loop).
// =====================================================================
__global__ __launch_bounds__(256)
void wmma_gemm_kernel(const float* __restrict__ A, const float* __restrict__ B,
                      const float* __restrict__ bias, float* __restrict__ Cout,
                      int M, int K, int Nc, int act)
{
    __shared__ _Float16 As[64][32];   // [m][k]               4 KB
    __shared__ _Float16 Bs[64][32];   // [n][k] (transposed)  4 KB

    const int tid  = threadIdx.x;
    const int lane = tid & 31;
    const int wave = tid >> 5;
    const int bm   = blockIdx.x * 64;
    const int bn   = blockIdx.y * 64;
    const int wrow = (wave & 3) << 4;   // 0,16,32,48
    const int wcol = (wave >> 2) << 5;  // 0,32

    const bool fullM = (bm + 64 <= M);
    const bool fullN = (bn + 64 <= Nc);
    const int  kFull = K & ~31;

    v8f acc0 = {};
    v8f acc1 = {};

    // fragment gather + 2 WMMAs (ISA wave32 16-bit layouts)
    auto mma_step = [&]() {
        // A 16x32: lane<16 -> K {0..7,16..23}; lane>=16 -> K {8..15,24..31}
        int arow  = wrow + (lane & 15);
        int abase = (lane < 16) ? 0 : 8;
        union { v16h v; v8h h[2]; } ua;
        ua.h[0] = *(const v8h*)&As[arow][abase];
        ua.h[1] = *(const v8h*)&As[arow][abase + 16];
        // B 32x16: lane<16 -> K {0..7,16..23}; lane>=16 -> K {8..15,24..31}; col = lane&15
        int bcol  = wcol + (lane & 15);
        int kbase = (lane < 16) ? 0 : 8;
        union { v16h v; v8h h[2]; } ub0, ub1;
        ub0.h[0] = *(const v8h*)&Bs[bcol][kbase];
        ub0.h[1] = *(const v8h*)&Bs[bcol][kbase + 16];
        ub1.h[0] = *(const v8h*)&Bs[bcol + 16][kbase];
        ub1.h[1] = *(const v8h*)&Bs[bcol + 16][kbase + 16];
        acc0 = __builtin_amdgcn_wmma_f32_16x16x32_f16(
                   false, ua.v, false, ub0.v, (short)0, acc0, false, false);
        acc1 = __builtin_amdgcn_wmma_f32_16x16x32_f16(
                   false, ua.v, false, ub1.v, (short)0, acc1, false, false);
    };

    // ---------------- main K loop: no K guards ----------------
    for (int k0 = 0; k0 < kFull; k0 += 32) {
        // stage A tile (64x32): float2 loads, row-clamped + select if edge tile
        #pragma unroll
        for (int i = 0; i < 4; ++i) {
            int idx = tid + i * 256;
            int r = idx >> 4, c = (idx & 15) << 1;
            int gr = bm + r;
            int cr = fullM ? gr : min(gr, M - 1);
            f2v g = *(const f2v*)&A[(long)cr * K + k0 + c];
            if (!fullM && gr >= M) { g.x = 0.0f; g.y = 0.0f; }
            h2 hv = { (_Float16)g.x, (_Float16)g.y };
            *(h2*)&As[r][c] = hv;                   // ds_store_b32
        }
        // stage B tile (32x64, transposed): col-clamped + selects if edge tile
        #pragma unroll
        for (int i = 0; i < 4; ++i) {
            int idx = tid + i * 256;
            int kk = idx >> 5, c = (idx & 31) << 1;
            int gc = bn + c;
            int cc = fullN ? gc : min(gc, Nc - 2);  // Nc is even in this pipeline
            f2v g = *(const f2v*)&B[(long)(k0 + kk) * Nc + cc];
            if (!fullN) {
                if (gc     >= Nc) g.x = 0.0f;
                if (gc + 1 >= Nc) g.y = 0.0f;
            }
            Bs[c][kk]     = (_Float16)g.x;
            Bs[c + 1][kk] = (_Float16)g.y;
        }
        // unconditional clamped prefetch of next K tile (global_prefetch_b8)
        {
            int pr = min(bm + (tid & 63), M - 1);
            int pk = min(k0 + 32 + (tid & 31), K - 1);
            __builtin_prefetch(&A[(long)pr * K + pk], 0, 1);
            __builtin_prefetch(&B[(long)pk * Nc + bn], 0, 1);
        }
        __syncthreads();
        mma_step();
        __syncthreads();
    }

    // ---------------- K remainder epilogue (runs at most once) ----------------
    if (kFull < K) {
        #pragma unroll
        for (int i = 0; i < 8; ++i) {
            int idx = tid + i * 256;
            int r = idx >> 5, c = idx & 31;
            int gr = bm + r, gc = kFull + c;
            float v = A[(long)min(gr, M - 1) * K + min(gc, K - 1)];
            As[r][c] = (_Float16)((gr < M && gc < K) ? v : 0.0f);
        }
        #pragma unroll
        for (int i = 0; i < 8; ++i) {
            int idx = tid + i * 256;
            int kk = idx >> 6, c = idx & 63;
            int gk = kFull + kk, gc = bn + c;
            float v = B[(long)min(gk, K - 1) * Nc + min(gc, Nc - 1)];
            Bs[c][kk] = (_Float16)((gk < K && gc < Nc) ? v : 0.0f);
        }
        __syncthreads();
        mma_step();
    }

    // ---- store D (VGPR r -> row r | r+8 by lane half; col = lane&15) ----
    int col0  = bn + wcol + (lane & 15);
    int col1  = col0 + 16;
    int rbase = bm + wrow + ((lane < 16) ? 0 : 8);
    if (fullM & fullN) {
        float bv0 = bias ? bias[col0] : 0.0f;
        float bv1 = bias ? bias[col1] : 0.0f;
        #pragma unroll
        for (int r = 0; r < 8; ++r) {
            int row = rbase + r;
            float v0 = acc0[r] + bv0, v1 = acc1[r] + bv1;
            if (act) { v0 = fmaxf(v0, 0.0f); v1 = fmaxf(v1, 0.0f); }
            Cout[(long)row * Nc + col0] = v0;
            Cout[(long)row * Nc + col1] = v1;
        }
    } else {
        float bv0 = (bias && col0 < Nc) ? bias[col0] : 0.0f;
        float bv1 = (bias && col1 < Nc) ? bias[col1] : 0.0f;
        #pragma unroll
        for (int r = 0; r < 8; ++r) {
            int row = rbase + r;
            if (row < M) {
                if (col0 < Nc) {
                    float v = acc0[r] + bv0;
                    if (act) v = fmaxf(v, 0.0f);
                    Cout[(long)row * Nc + col0] = v;
                }
                if (col1 < Nc) {
                    float v = acc1[r] + bv1;
                    if (act) v = fmaxf(v, 0.0f);
                    Cout[(long)row * Nc + col1] = v;
                }
            }
        }
    }
}

// =====================================================================
// GAT helpers
// =====================================================================
__device__ __forceinline__ void atomicMaxFloat(float* addr, float val)
{
    if (__float_as_int(val) >= 0)
        atomicMax((int*)addr, __float_as_int(val));
    else
        atomicMin((unsigned int*)addr, __float_as_uint(val));
}

__global__ void gat_init_kernel(float* __restrict__ res, const float* __restrict__ gbias,
                                float* __restrict__ segmax, float* __restrict__ denom)
{
    int idx = blockIdx.x * blockDim.x + threadIdx.x;
    if (idx < N_ * HC_) res[idx] = gbias[idx & (HC_ - 1)];
    if (idx < N_ * H_) { segmax[idx] = -__builtin_inff(); denom[idx] = 0.0f; }
}

__global__ void edge_logits_kernel(const int* __restrict__ ei,
                                   const float* __restrict__ xl, const float* __restrict__ xr,
                                   const float* __restrict__ att,
                                   float* __restrict__ logits, float* __restrict__ segmax)
{
    int idx = blockIdx.x * blockDim.x + threadIdx.x;
    if (idx >= NE_ * H_) return;
    int e = idx >> 3, hh = idx & 7;
    int src = (e < E_) ? ei[e]      : (e - E_);
    int dst = (e < E_) ? ei[E_ + e] : (e - E_);
    // all three row bases are 256-B aligned (multiples of 64 floats)
    const float4* pl = (const float4*)(xl + (long)src * HC_ + hh * C_);
    const float4* pr = (const float4*)(xr + (long)dst * HC_ + hh * C_);
    const float4* pa = (const float4*)(att + hh * C_);
    float s = 0.0f;
    #pragma unroll 4
    for (int c = 0; c < C_ / 4; ++c) {
        float4 a = pl[c], b = pr[c], w = pa[c];
        float v0 = a.x + b.x, v1 = a.y + b.y, v2 = a.z + b.z, v3 = a.w + b.w;
        v0 = (v0 > 0.0f) ? v0 : v0 * NEG_SLOPE;
        v1 = (v1 > 0.0f) ? v1 : v1 * NEG_SLOPE;
        v2 = (v2 > 0.0f) ? v2 : v2 * NEG_SLOPE;
        v3 = (v3 > 0.0f) ? v3 : v3 * NEG_SLOPE;
        s += v0 * w.x + v1 * w.y + v2 * w.z + v3 * w.w;
    }
    logits[idx] = s;
    atomicMaxFloat(&segmax[dst * H_ + hh], s);
}

__global__ void edge_exp_kernel(const int* __restrict__ ei,
                                float* __restrict__ logits,
                                const float* __restrict__ segmax,
                                float* __restrict__ denom)
{
    int idx = blockIdx.x * blockDim.x + threadIdx.x;
    if (idx >= NE_ * H_) return;
    int e = idx >> 3, hh = idx & 7;
    int dst = (e < E_) ? ei[E_ + e] : (e - E_);
    float ex = expf(logits[idx] - segmax[dst * H_ + hh]);
    logits[idx] = ex;
    atomicAdd(&denom[dst * H_ + hh], ex);
}

__global__ void edge_scatter_kernel(const int* __restrict__ ei,
                                    const float* __restrict__ logits,
                                    const float* __restrict__ denom,
                                    const float* __restrict__ xl,
                                    float* __restrict__ res)
{
    int idx = blockIdx.x * blockDim.x + threadIdx.x;
    if (idx >= NE_ * H_) return;
    int e = idx >> 3, hh = idx & 7;
    int src = (e < E_) ? ei[e]      : (e - E_);
    int dst = (e < E_) ? ei[E_ + e] : (e - E_);
    float alpha = logits[idx] / (denom[dst * H_ + hh] + 1e-16f);
    const float4* pl = (const float4*)(xl + (long)src * HC_ + hh * C_);
    float*        pd = res + (long)dst * HC_ + hh * C_;
    #pragma unroll 4
    for (int c = 0; c < C_ / 4; ++c) {
        float4 a = pl[c];
        atomicAdd(&pd[c * 4 + 0], alpha * a.x);
        atomicAdd(&pd[c * 4 + 1], alpha * a.y);
        atomicAdd(&pd[c * 4 + 2], alpha * a.z);
        atomicAdd(&pd[c * 4 + 3], alpha * a.w);
    }
}

// =====================================================================
// CNN branches: res (N,1,8,64) -> cat (N, 1242), relu fused
// =====================================================================
__global__ void cnn_kernel(const float* __restrict__ res,
                           const float* __restrict__ w1,  const float* __restrict__ b1,
                           const float* __restrict__ w4,  const float* __restrict__ b4,
                           const float* __restrict__ w16, const float* __restrict__ b16,
                           const float* __restrict__ w32, const float* __restrict__ b32,
                           float* __restrict__ cat)
{
    int f = blockIdx.x * blockDim.x + threadIdx.x;
    int n = blockIdx.y;
    if (f >= F_) return;

    const float* w; const float* bb; int k, wdt, off;
    if      (f < 384)  { k = 1;  wdt = 64; off = 0;    w = w1;  bb = b1;  }
    else if (f < 750)  { k = 4;  wdt = 61; off = 384;  w = w4;  bb = b4;  }
    else if (f < 1044) { k = 16; wdt = 49; off = 750;  w = w16; bb = b16; }
    else               { k = 32; wdt = 33; off = 1044; w = w32; bb = b32; }

    int local = f - off;
    int o = local / wdt, j = local % wdt;
    const float* xp = res + (long)n * HC_;
    float s = bb[o];
    for (int hh = 0; hh < H_; ++hh) {
        const float* row = xp + hh * C_ + j;
        const float* wr  = w + (o * H_ + hh) * k;
        for (int t = 0; t < k; ++t)
            s += row[t] * wr[t];
    }
    cat[(long)n * F_ + f] = fmaxf(s, 0.0f);
}

// =====================================================================
// pairs = rna[i] * dis[j] (716 MB stream -> nontemporal f2v), + labels
// =====================================================================
__global__ __launch_bounds__(256)
void pairs_kernel(const float* __restrict__ feat, const float* __restrict__ rel,
                  float* __restrict__ out)
{
    int p = blockIdx.x;                 // 0 .. R*D-1
    int i = p / D_, j = p % D_;
    // F_ = 1242 is even: rows are 8-B aligned -> f2v everywhere
    const f2v* ra = (const f2v*)(feat + (long)i * F_);
    const f2v* db = (const f2v*)(feat + (long)(R_ + j) * F_);
    f2v*       po = (f2v*)(out + (size_t)p * F_);
    for (int f = threadIdx.x; f < F_ / 2; f += blockDim.x) {
        f2v a = ra[f], b = db[f];
        f2v v = a * b;
        __builtin_nontemporal_store(v, &po[f]);
    }
    if (threadIdx.x == 0)
        out[(size_t)R_ * D_ * F_ + p] = rel[p];   // labels
}

// =====================================================================
extern "C" void kernel_launch(void* const* d_in, const int* in_sizes, int n_in,
                              void* d_out, int out_size, void* d_ws, size_t ws_size,
                              hipStream_t stream)
{
    const int*   ei    = (const int*)  d_in[0];
    const float* met   = (const float*)d_in[1];
    const float* dis   = (const float*)d_in[2];
    const float* rel   = (const float*)d_in[3];
    const float* Wrna  = (const float*)d_in[4];
    const float* Wdis  = (const float*)d_in[5];
    const float* Wl    = (const float*)d_in[6];
    const float* bl    = (const float*)d_in[7];
    const float* Wr    = (const float*)d_in[8];
    const float* br    = (const float*)d_in[9];
    const float* att   = (const float*)d_in[10];
    const float* gbias = (const float*)d_in[11];
    const float* w1    = (const float*)d_in[12];
    const float* b1    = (const float*)d_in[13];
    const float* w4    = (const float*)d_in[14];
    const float* b4    = (const float*)d_in[15];
    const float* w16   = (const float*)d_in[16];
    const float* b16   = (const float*)d_in[17];
    const float* w32   = (const float*)d_in[18];
    const float* b32   = (const float*)d_in[19];
    const float* Wfc   = (const float*)d_in[20];
    const float* bfc   = (const float*)d_in[21];
    float* out = (float*)d_out;

    // workspace layout (floats): ~13.8 MB total
    float* ws     = (float*)d_ws;
    float* h      = ws;  ws += N_ * C_;     // 840x64
    float* xl     = ws;  ws += N_ * HC_;    // 840x512
    float* xr     = ws;  ws += N_ * HC_;
    float* logits = ws;  ws += NE_ * H_;    // 20840x8 (reused as ex)
    float* segmax = ws;  ws += N_ * H_;
    float* denom  = ws;  ws += N_ * H_;
    float* res    = ws;  ws += N_ * HC_;
    float* cat    = ws;  ws += N_ * F_;     // 840x1242
    float* feat   = ws;  ws += N_ * F_;

    const dim3 blk(256);
    auto cdiv = [](int a, int b) { return (a + b - 1) / b; };

    // 1) h = [met @ W_rna ; dis @ W_dis]
    wmma_gemm_kernel<<<dim3(cdiv(R_, 64), cdiv(C_, 64)), blk, 0, stream>>>(
        met, Wrna, nullptr, h, R_, R_, C_, 0);
    wmma_gemm_kernel<<<dim3(cdiv(D_, 64), cdiv(C_, 64)), blk, 0, stream>>>(
        dis, Wdis, nullptr, h + (long)R_ * C_, D_, D_, C_, 0);

    // 2) xl / xr projections
    wmma_gemm_kernel<<<dim3(cdiv(N_, 64), cdiv(HC_, 64)), blk, 0, stream>>>(
        h, Wl, bl, xl, N_, C_, HC_, 0);
    wmma_gemm_kernel<<<dim3(cdiv(N_, 64), cdiv(HC_, 64)), blk, 0, stream>>>(
        h, Wr, br, xr, N_, C_, HC_, 0);

    // 3) GATv2 edge softmax + scatter
    gat_init_kernel<<<cdiv(N_ * HC_, 256), blk, 0, stream>>>(res, gbias, segmax, denom);
    int egrid = cdiv(NE_ * H_, 256);
    edge_logits_kernel <<<egrid, blk, 0, stream>>>(ei, xl, xr, att, logits, segmax);
    edge_exp_kernel    <<<egrid, blk, 0, stream>>>(ei, logits, segmax, denom);
    edge_scatter_kernel<<<egrid, blk, 0, stream>>>(ei, logits, denom, xl, res);

    // 4) CNN branches -> cat
    cnn_kernel<<<dim3(cdiv(F_, 256), N_), blk, 0, stream>>>(
        res, w1, b1, w4, b4, w16, b16, w32, b32, cat);

    // 5) feat = relu(cat @ W_fc + b_fc)   (the big 840x1242x1242 GEMM)
    wmma_gemm_kernel<<<dim3(cdiv(N_, 64), cdiv(F_, 64)), blk, 0, stream>>>(
        cat, Wfc, bfc, feat, N_, F_, F_, 1);

    // 6) pairs outer product + labels (HBM-write bound; nontemporal)
    pairs_kernel<<<R_ * D_, blk, 0, stream>>>(feat, rel, out);
}
